// GATSimple_2001454760655
// MI455X (gfx1250) — matrix operand order repaired and checked by reading, not compiled, not run on hardware
//
#include <hip/hip_runtime.h>
#include <hip/hip_bf16.h>

#define GAT_N    100000
#define GAT_FIN  256
#define GAT_FOUT 64
#define GAT_E    3200000
#define GAT_TOT  (GAT_E + GAT_N)   // edges + self loops
#define NEG_SLOPE 0.2f

typedef __attribute__((ext_vector_type(16))) __bf16 v16bf;
typedef __attribute__((ext_vector_type(8)))  float  v8f;

// ---------- ordered-float encoding for atomic max ----------
__device__ __forceinline__ unsigned enc_f(float f) {
    unsigned u = __float_as_uint(f);
    return (u & 0x80000000u) ? ~u : (u | 0x80000000u);
}
__device__ __forceinline__ float dec_f(unsigned u) {
    return (u & 0x80000000u) ? __uint_as_float(u & 0x7FFFFFFFu)
                             : __uint_as_float(~u);
}

// ---------- pack W (256x64 f32, row major) into WMMA B-operand layout ----------
// Entry t = (kt*4 + nt)*32 + lane holds 16 bf16: element i -> W[K][col],
// K = kt*32 + (lane<16 ? 0 : 8) + (i<8 ? i : 8+i), col = nt*16 + (lane&15).
__global__ void gat_pack_w(const float* __restrict__ W, __bf16* __restrict__ wpack) {
    int t = blockIdx.x * blockDim.x + threadIdx.x;   // 0..1023
    if (t >= 8 * 4 * 32) return;
    int kt   = t >> 7;
    int nt   = (t >> 5) & 3;
    int lane = t & 31;
    int col  = nt * 16 + (lane & 15);
    int kbase = kt * 32 + ((lane >> 4) << 3);        // 0 or +8
    __bf16* dst = wpack + (size_t)t * 16;
#pragma unroll
    for (int i = 0; i < 16; ++i) {
        dst[i] = (__bf16)W[(size_t)(kbase + i + ((i >> 3) << 3)) * GAT_FOUT + col];
    }
}

// ---------- init per-node accumulators ----------
__global__ void gat_init_nodes(unsigned* __restrict__ emax, float* __restrict__ denom) {
    int n = blockIdx.x * blockDim.x + threadIdx.x;
    if (n >= GAT_N) return;
    emax[n]  = 0x007FFFFFu;   // enc(-inf)
    denom[n] = 0.0f;
}

// ---------- init output with bias ----------
__global__ void gat_init_out(float* __restrict__ out, const float* __restrict__ bias) {
    int idx = blockIdx.x * blockDim.x + threadIdx.x;
    if (idx >= GAT_N * GAT_FOUT) return;
    out[idx] = bias[idx & (GAT_FOUT - 1)];
}

// ---------- h = x @ W via v_wmma_f32_16x16x32_bf16 ----------
// One wave computes a 16x64 tile: 4 accumulators of 16x16, K-loop of 8 steps.
__global__ void gat_gemm_wmma(const float* __restrict__ x,
                              const __bf16* __restrict__ wpack,
                              float* __restrict__ h) {
    const int lane = threadIdx.x & 31;
    const int wv   = threadIdx.x >> 5;
    const int tile = blockIdx.x * 8 + wv;
    if (tile >= GAT_N / 16) return;                  // 6250 row tiles

    const int row0  = tile * 16;
    const int m     = lane & 15;                     // A-row / D-column lane id
    const int khalf = (lane >> 4) << 3;              // 0 or 8

    const float* xrow = x + (size_t)(row0 + m) * GAT_FIN;

    v8f acc[4];
#pragma unroll
    for (int i = 0; i < 4; ++i)
        acc[i] = (v8f){0.f, 0.f, 0.f, 0.f, 0.f, 0.f, 0.f, 0.f};

#pragma unroll
    for (int kt = 0; kt < 8; ++kt) {
        const int k0 = kt * 32 + khalf;
        float4 lo0 = *(const float4*)(xrow + k0);
        float4 lo1 = *(const float4*)(xrow + k0 + 4);
        float4 hi0 = *(const float4*)(xrow + k0 + 16);
        float4 hi1 = *(const float4*)(xrow + k0 + 20);

        v16bf a;
        a[0]  = (__bf16)lo0.x; a[1]  = (__bf16)lo0.y;
        a[2]  = (__bf16)lo0.z; a[3]  = (__bf16)lo0.w;
        a[4]  = (__bf16)lo1.x; a[5]  = (__bf16)lo1.y;
        a[6]  = (__bf16)lo1.z; a[7]  = (__bf16)lo1.w;
        a[8]  = (__bf16)hi0.x; a[9]  = (__bf16)hi0.y;
        a[10] = (__bf16)hi0.z; a[11] = (__bf16)hi0.w;
        a[12] = (__bf16)hi1.x; a[13] = (__bf16)hi1.y;
        a[14] = (__bf16)hi1.z; a[15] = (__bf16)hi1.w;

#pragma unroll
        for (int nt = 0; nt < 4; ++nt) {
            const v16bf b =
                *(const v16bf*)(wpack + ((size_t)(kt * 4 + nt) * 32 + lane) * 16);
            acc[nt] = __builtin_amdgcn_wmma_f32_16x16x32_bf16(
                false, a, false, b, (short)0, acc[nt], false, false);
        }
    }

    // D layout: lanes 0-15 hold M=r (VGPR r), lanes 16-31 hold M=r+8; N = lane&15
    const int mrow = (lane >> 4) << 3;
#pragma unroll
    for (int nt = 0; nt < 4; ++nt) {
#pragma unroll
        for (int r = 0; r < 8; ++r) {
            h[(size_t)(row0 + mrow + r) * GAT_FOUT + nt * 16 + m] = acc[nt][r];
        }
    }
}

// ---------- per-node attention halves: a_src = h.att_src, a_dst = h.att_dst ----------
__global__ void gat_attn_scores(const float* __restrict__ h,
                                const float* __restrict__ att_src,
                                const float* __restrict__ att_dst,
                                float* __restrict__ a_src,
                                float* __restrict__ a_dst) {
    int n = blockIdx.x * blockDim.x + threadIdx.x;
    if (n >= GAT_N) return;
    const float* hr = h + (size_t)n * GAT_FOUT;
    float s = 0.f, d = 0.f;
#pragma unroll
    for (int i = 0; i < GAT_FOUT; i += 4) {
        float4 hv = *(const float4*)(hr + i);
        float4 as = *(const float4*)(att_src + i);
        float4 ad = *(const float4*)(att_dst + i);
        s += hv.x * as.x + hv.y * as.y + hv.z * as.z + hv.w * as.w;
        d += hv.x * ad.x + hv.y * ad.y + hv.z * ad.z + hv.w * ad.w;
    }
    a_src[n] = s;
    a_dst[n] = d;
}

// ---------- pass 1: leaky-relu score + segment max (ordered-uint atomic max) ----------
__global__ void gat_edge_max(const int* __restrict__ ei,
                             const float* __restrict__ a_src,
                             const float* __restrict__ a_dst,
                             float* __restrict__ eval,
                             unsigned* __restrict__ emax) {
    int e = blockIdx.x * blockDim.x + threadIdx.x;
    if (e >= GAT_TOT) return;
    int s, d;
    if (e < GAT_E) { s = ei[e]; d = ei[GAT_E + e]; }
    else           { s = d = e - GAT_E; }
    float v  = a_src[s] + a_dst[d];
    float ev = v > 0.f ? v : NEG_SLOPE * v;
    eval[e] = ev;
    atomicMax(emax + d, enc_f(ev));
}

// ---------- pass 2: p = exp(e - max[dst]); denom[dst] += p ----------
__global__ void gat_edge_exp(const int* __restrict__ ei,
                             const float* __restrict__ eval,
                             const unsigned* __restrict__ emax,
                             float* __restrict__ p,
                             float* __restrict__ denom) {
    int e = blockIdx.x * blockDim.x + threadIdx.x;
    if (e >= GAT_TOT) return;
    int d;
    if (e < GAT_E) d = ei[GAT_E + e];
    else           d = e - GAT_E;
    float em = dec_f(emax[d]);
    float pe = __expf(eval[e] - em);
    p[e] = pe;
    atomicAdd(denom + d, pe);
}

// ---------- pass 3: out[dst] += alpha * h[src]; one wave32 per edge ----------
__global__ void gat_edge_aggr(const int* __restrict__ ei,
                              const float* __restrict__ h,
                              const float* __restrict__ p,
                              const float* __restrict__ denom,
                              float* __restrict__ out) {
    int wave = (blockIdx.x * blockDim.x + threadIdx.x) >> 5;
    int lane = threadIdx.x & 31;
    if (wave >= GAT_TOT) return;
    int e = wave;
    int s, d;
    if (e < GAT_E) { s = ei[e]; d = ei[GAT_E + e]; }
    else           { s = d = e - GAT_E; }
    float alpha = p[e] / (denom[d] + 1e-16f);          // uniform across wave
    const float2 hv = *(const float2*)(h + (size_t)s * GAT_FOUT + lane * 2);
    float* o = out + (size_t)d * GAT_FOUT + lane * 2;
    atomicAdd(o,     alpha * hv.x);
    atomicAdd(o + 1, alpha * hv.y);
}

static inline size_t align256(size_t v) { return (v + 255) & ~(size_t)255; }

extern "C" void kernel_launch(void* const* d_in, const int* in_sizes, int n_in,
                              void* d_out, int out_size, void* d_ws, size_t ws_size,
                              hipStream_t stream) {
    const float* x       = (const float*)d_in[0];
    const int*   ei      = (const int*)d_in[1];
    const float* W       = (const float*)d_in[2];
    const float* att_src = (const float*)d_in[3];
    const float* att_dst = (const float*)d_in[4];
    const float* bias    = (const float*)d_in[5];
    float*       out     = (float*)d_out;

    // carve workspace
    char* ws = (char*)d_ws;
    size_t off = 0;
    __bf16* wpack = (__bf16*)(ws + off); off += align256(8 * 4 * 32 * 16 * sizeof(__bf16));
    float* h      = (float*)(ws + off);  off += align256((size_t)GAT_N * GAT_FOUT * sizeof(float));
    float* a_src  = (float*)(ws + off);  off += align256((size_t)GAT_N * sizeof(float));
    float* a_dst  = (float*)(ws + off);  off += align256((size_t)GAT_N * sizeof(float));
    unsigned* emax= (unsigned*)(ws + off); off += align256((size_t)GAT_N * sizeof(unsigned));
    float* denom  = (float*)(ws + off);  off += align256((size_t)GAT_N * sizeof(float));
    float* eval   = (float*)(ws + off);  off += align256((size_t)GAT_TOT * sizeof(float));
    float* p      = (float*)(ws + off);  off += align256((size_t)GAT_TOT * sizeof(float));
    (void)ws_size; (void)in_sizes; (void)n_in; (void)out_size;

    const int B = 256;

    gat_pack_w<<<4, B, 0, stream>>>(W, wpack);
    gat_init_nodes<<<(GAT_N + B - 1) / B, B, 0, stream>>>(emax, denom);
    gat_init_out<<<((GAT_N * GAT_FOUT) + B - 1) / B, B, 0, stream>>>(out, bias);

    // 6250 row-tiles of 16, 8 waves per 256-thread block
    gat_gemm_wmma<<<(GAT_N / 16 + 7) / 8, B, 0, stream>>>(x, wpack, h);

    gat_attn_scores<<<(GAT_N + B - 1) / B, B, 0, stream>>>(h, att_src, att_dst, a_src, a_dst);

    gat_edge_max<<<(GAT_TOT + B - 1) / B, B, 0, stream>>>(ei, a_src, a_dst, eval, emax);
    gat_edge_exp<<<(GAT_TOT + B - 1) / B, B, 0, stream>>>(ei, eval, emax, p, denom);

    // one wave32 per edge: 8 edges per 256-thread block
    gat_edge_aggr<<<(GAT_TOT + 7) / 8, B, 0, stream>>>(ei, h, p, denom, out);
}
